// GraphAttentionLayer_57260503990856
// MI455X (gfx1250) — matrix operand order repaired and checked by reading, not compiled, not run on hardware
//
#include <hip/hip_runtime.h>
#include <hip/hip_fp16.h>

typedef _Float16 v16h  __attribute__((ext_vector_type(16)));
typedef __bf16   v16bf __attribute__((ext_vector_type(16)));
typedef float    v8f   __attribute__((ext_vector_type(8)));
typedef unsigned int v4u __attribute__((ext_vector_type(4)));
typedef int      v4i   __attribute__((ext_vector_type(4)));
typedef int      v8i   __attribute__((ext_vector_type(8)));

union HPack {                 // 16 x f16  <-> 2 x b128
    v16h v;
    uint4 q[2];
    _Float16 h[16];
};
union BfPack {                // 16 x bf16 <-> 2 x b128
    v16bf v;
    uint4 q[2];
    unsigned short s[16];
};
union H8Pack {                // 8 x f16 <-> 1 x b128
    _Float16 h[8];
    uint4 q;
};

__device__ __forceinline__ unsigned short f32_to_bf16(float f) {
    unsigned int x = __float_as_uint(f);
    unsigned int r = (x + 0x7FFFu + ((x >> 16) & 1u)) >> 16;   // RNE
    return (unsigned short)r;
}

#if defined(__has_builtin)
#if __has_builtin(__builtin_amdgcn_tensor_load_to_lds) && \
    __has_builtin(__builtin_amdgcn_s_wait_tensorcnt)
#define HAVE_TDM 1
#endif
#endif

#ifdef HAVE_TDM
// ---------------------------------------------------------------------------
// TDM: DMA hT[:, jc:jc+32] (256 rows x 64B, pitch 16KB) into LDS, inserting
// 16B of padding after every 64B so the LDS image has an 80B row stride
// (conflict-free for the WMMA B-operand reads). D# per CDNA5 ISA.
// ---------------------------------------------------------------------------
__device__ __forceinline__ void tdm_load_chunk(const _Float16* gsrc,
                                               unsigned lds_off) {
    unsigned long long ga = (unsigned long long)(uintptr_t)gsrc;
    v4u g0 = { 1u,                                    // count=1 (valid D#)
               lds_off,                                // lds_addr
               (unsigned)(ga & 0xFFFFFFFFull),        // global_addr[31:0]
               (unsigned)((ga >> 32) & 0x01FFFFFFull) // global_addr[56:32]
                   | (2u << 30) };                    // type=2 ("image")
    v8i g1 = { (int)((1u << 16)        // data_size = 2 bytes
                   | (1u << 20)        // pad_enable
                   | (3u << 22)        // pad_interval: 16 DWORDs (64B)
                   | (3u << 25)),      // pad_amount:   4 DWORDs (16B)
               (int)(0x2000u << 16),   // tensor_dim0 = 8192 (lo16 -> [31:16])
               (int)(256u << 16),      // dim0 hi=0 | tensor_dim1=256 lo16
               (int)(32u << 16),       // dim1 hi=0 | tile_dim0 = 32
               256,                    // tile_dim1 = 256 | tile_dim2 = 0
               8192,                   // tensor_dim0_stride lo32 (elements)
               0, 0 };                 // stride hi | dim1_stride = 0
    v4i gz = { 0, 0, 0, 0 };
#if __clang_major__ >= 23
    v8i gz8 = { 0, 0, 0, 0, 0, 0, 0, 0 };
    __builtin_amdgcn_tensor_load_to_lds(g0, g1, gz, gz, gz8, 0);
#else
    __builtin_amdgcn_tensor_load_to_lds(g0, g1, gz, gz, 0);
#endif
}
#endif // HAVE_TDM

// ---------------------------------------------------------------------------
// Kernel 0: W (512x256 f32, row-major) -> Wt (256x512 bf16, "col-major")
// so WMMA B-operand reads are 16 contiguous bf16 per lane (2 x b128).
// ---------------------------------------------------------------------------
__global__ void __launch_bounds__(256) k_wt(const float* __restrict__ W,
                                            unsigned short* __restrict__ Wt) {
    int idx = blockIdx.x * 256 + threadIdx.x;   // 131072 total
    int n = idx >> 9;          // 0..255  (output column of h)
    int k = idx & 511;         // 0..511
    Wt[idx] = f32_to_bf16(W[k * 256 + n]);
}

// ---------------------------------------------------------------------------
// Kernel 1: h = x @ W via v_wmma_f32_16x16x32_bf16.
// Each wave: one 16-row x 256-col tile of h, K-loop over 512.
// Fused epilogues: f = h@a1, g = h@a2 (shfl reductions), transposed f16
// store hT[col][row] (contiguous b128 per lane thanks to the D layout).
// ---------------------------------------------------------------------------
__global__ void __launch_bounds__(256) k_h(const float* __restrict__ x,
                                           const unsigned short* __restrict__ Wt,
                                           const float* __restrict__ a,
                                           _Float16* __restrict__ hT,
                                           float* __restrict__ f,
                                           float* __restrict__ g) {
    const int tid  = threadIdx.x;
    const int wave = tid >> 5;
    const int lane = tid & 31;
    const int m    = lane & 15;       // A-matrix row of this lane
    const int hh   = lane >> 4;       // lane half (0: lanes 0-15, 1: 16-31)
    const int row0 = (blockIdx.x * 8 + wave) * 16;

    v8f acc[16];
    const v8f zacc = {0.f, 0.f, 0.f, 0.f, 0.f, 0.f, 0.f, 0.f};
#pragma unroll
    for (int nt = 0; nt < 16; ++nt) acc[nt] = zacc;

    const float* xr = x + (size_t)(row0 + m) * 512;

    for (int kb = 0; kb < 512; kb += 32) {
        // --- A operand: 16 bf16 per lane.
        // elements 0..7  = K = kb + 16*hh + (0..7)
        // elements 8..15 = K = kb + 16*hh + 8 + (0..7)
        const float4* xa = (const float4*)(xr + kb + hh * 16);
        const float4* xb = (const float4*)(xr + kb + hh * 16 + 8);
        float4 x0 = xa[0], x1 = xa[1], x2 = xb[0], x3 = xb[1];
        BfPack A;
        A.s[0]  = f32_to_bf16(x0.x); A.s[1]  = f32_to_bf16(x0.y);
        A.s[2]  = f32_to_bf16(x0.z); A.s[3]  = f32_to_bf16(x0.w);
        A.s[4]  = f32_to_bf16(x1.x); A.s[5]  = f32_to_bf16(x1.y);
        A.s[6]  = f32_to_bf16(x1.z); A.s[7]  = f32_to_bf16(x1.w);
        A.s[8]  = f32_to_bf16(x2.x); A.s[9]  = f32_to_bf16(x2.y);
        A.s[10] = f32_to_bf16(x2.z); A.s[11] = f32_to_bf16(x2.w);
        A.s[12] = f32_to_bf16(x3.x); A.s[13] = f32_to_bf16(x3.y);
        A.s[14] = f32_to_bf16(x3.z); A.s[15] = f32_to_bf16(x3.w);

#pragma unroll
        for (int nt = 0; nt < 16; ++nt) {
            // B operand: lane column n = nt*16 + m, K = kb + 16*hh + (0..15)
            int col = nt * 16 + m;
            const uint4* bp = (const uint4*)(Wt + (size_t)col * 512 + kb + hh * 16);
            BfPack B;
            B.q[0] = bp[0];
            B.q[1] = bp[1];
            acc[nt] = __builtin_amdgcn_wmma_f32_16x16x32_bf16(
                false, A.v, false, B.v, (short)0, acc[nt], false, false);
        }
    }

    // --- fused f = h@a1, g = h@a2.
    // D layout: lane l, vgpr r -> (row = r + 8*hh, col = nt*16 + m)
#pragma unroll
    for (int r = 0; r < 8; ++r) {
        float pf = 0.f, pg = 0.f;
#pragma unroll
        for (int nt = 0; nt < 16; ++nt) {
            int n = nt * 16 + m;
            float hv = acc[nt][r];
            pf += hv * a[n];
            pg += hv * a[256 + n];
        }
#pragma unroll
        for (int o = 1; o < 16; o <<= 1) {       // reduce within each 16-lane half
            pf += __shfl_xor(pf, o, 32);
            pg += __shfl_xor(pg, o, 32);
        }
        if (m == 0) {
            int row = row0 + r + 8 * hh;
            f[row] = pf;
            g[row] = pg;
        }
    }

    // --- transposed f16 store: hT[col][row]; 8 consecutive rows per lane.
#pragma unroll
    for (int nt = 0; nt < 16; ++nt) {
        H8Pack hp;
#pragma unroll
        for (int r = 0; r < 8; ++r) hp.h[r] = (_Float16)acc[nt][r];
        int col = nt * 16 + m;
        *(uint4*)(hT + (size_t)col * 8192 + row0 + 8 * hh) = hp.q;
    }
}

// ---------------------------------------------------------------------------
// Kernel 2: gmax = max_j g[j]   (relu monotone => row max m_i = relu(f_i+gmax))
// ---------------------------------------------------------------------------
__global__ void __launch_bounds__(256) k_gmax(const float* __restrict__ g,
                                              float* __restrict__ gmax) {
    __shared__ float red[256];
    float mx = -3.4e38f;
    for (int i = threadIdx.x; i < 8192; i += 256) mx = fmaxf(mx, g[i]);
    red[threadIdx.x] = mx;
    __syncthreads();
    for (int s = 128; s > 0; s >>= 1) {
        if (threadIdx.x < (unsigned)s)
            red[threadIdx.x] = fmaxf(red[threadIdx.x], red[threadIdx.x + s]);
        __syncthreads();
    }
    if (threadIdx.x == 0) *gmax = red[0];
}

// ---------------------------------------------------------------------------
// Kernel 3: fused softmax(relu(f_i + g_j)) @ h, + ELU.
// Block = 8 waves x 16 rows = 128 rows. j processed in chunks of 32.
// Double-buffered LDS staging via the Tensor Data Mover (wave 0 issues the
// DMA for chunk i+1, s_wait_tensorcnt guarantees chunk i, barrier publishes),
// so all global->LDS traffic overlaps the WMMA stream.
// ---------------------------------------------------------------------------
__global__ void __launch_bounds__(256) k_attn(const _Float16* __restrict__ hT,
                                              const float* __restrict__ f,
                                              const float* __restrict__ g,
                                              const float* __restrict__ gmax,
                                              float* __restrict__ out) {
    __shared__ __align__(16) _Float16 ldsB[2][256 * 40];  // 2 x 20 KB

    const int tid  = threadIdx.x;
    const int wave = tid >> 5;
    const int lane = tid & 31;
    const int m    = lane & 15;
    const int hh   = lane >> 4;
    const int row0 = (blockIdx.x * 8 + wave) * 16;
    const int NIT  = 256;                 // 8192 / 32

    const float fm = f[row0 + m];
    const float mi = fmaxf(fm + *gmax, 0.f);   // exact row max of relu(f+g)

    v8f acc[16];
    const v8f zacc = {0.f, 0.f, 0.f, 0.f, 0.f, 0.f, 0.f, 0.f};
#pragma unroll
    for (int nt = 0; nt < 16; ++nt) acc[nt] = zacc;
    float sacc = 0.f;

#ifdef HAVE_TDM
    if (wave == 0)                        // prologue: chunk 0 -> buffer 0
        tdm_load_chunk(hT, (unsigned)(uintptr_t)&ldsB[0][0]);
#endif

    for (int it = 0; it < NIT; ++it) {
        const int jc = it * 32;
        __syncthreads();   // everyone done reading buffer (it+1)&1: safe to fill
#ifdef HAVE_TDM
        if (wave == 0) {
            if (it + 1 < NIT) {
                // DMA next chunk into the other buffer, then wait until the
                // older outstanding TDM op (chunk `it`, in-order) completed.
                tdm_load_chunk(hT + (size_t)(jc + 32),
                               (unsigned)(uintptr_t)&ldsB[(it + 1) & 1][0]);
                __builtin_amdgcn_s_wait_tensorcnt(1);
            } else {
                __builtin_amdgcn_s_wait_tensorcnt(0);
            }
        }
#else
        {   // fallback staging: one column (64B) per thread
            const uint4* src = (const uint4*)(hT + (size_t)tid * 8192 + jc);
            uint4* dst = (uint4*)(&ldsB[it & 1][0] + tid * 40);
            dst[0] = src[0]; dst[1] = src[1]; dst[2] = src[2]; dst[3] = src[3];
        }
#endif
        __syncthreads();   // buffer it&1 now visible to all waves
        const _Float16* buf = &ldsB[it & 1][0];

        // --- A operand: p = exp(relu(f_m + g_j) - m_i), f16, per the
        // contiguous k order used here (elements 0..15: k = 16*hh+0..15, matching B)
        const float4* g0p = (const float4*)(g + jc + hh * 16);
        const float4* g1p = (const float4*)(g + jc + hh * 16 + 8);
        float4 g0 = g0p[0], g1 = g0p[1], g2 = g1p[0], g3 = g1p[1];
        float pv[16] = {g0.x, g0.y, g0.z, g0.w, g1.x, g1.y, g1.z, g1.w,
                        g2.x, g2.y, g2.z, g2.w, g3.x, g3.y, g3.z, g3.w};
        HPack A;
#pragma unroll
        for (int e = 0; e < 16; ++e) {
            float p = __expf(fmaxf(fm + pv[e], 0.f) - mi);   // in (0, 1]
            sacc += p;
            A.h[e] = (_Float16)p;
        }

#pragma unroll
        for (int nt = 0; nt < 16; ++nt) {
            int col = nt * 16 + m;
            const uint4* bp = (const uint4*)(buf + col * 40 + hh * 16);
            HPack B;
            B.q[0] = bp[0];
            B.q[1] = bp[1];
            acc[nt] = __builtin_amdgcn_wmma_f32_16x16x32_f16(
                false, A.v, false, B.v, (short)0, acc[nt], false, false);
        }
    }

    // --- softmax denominator per row, then divide + ELU + store.
    float stot = sacc + __shfl_xor(sacc, 16, 32);   // lane L holds s for row L&15
#pragma unroll
    for (int r = 0; r < 8; ++r) {
        float s  = __shfl(stot, r + 8 * hh, 32);
        int row  = row0 + r + 8 * hh;
#pragma unroll
        for (int nt = 0; nt < 16; ++nt) {
            float v = acc[nt][r] / s;
            out[(size_t)row * 256 + nt * 16 + m] = (v > 0.f) ? v : expm1f(v);
        }
    }
}

// ---------------------------------------------------------------------------
extern "C" void kernel_launch(void* const* d_in, const int* in_sizes, int n_in,
                              void* d_out, int out_size, void* d_ws, size_t ws_size,
                              hipStream_t stream) {
    (void)in_sizes; (void)n_in; (void)out_size; (void)ws_size;
    const float* x = (const float*)d_in[0];   // 8192 x 512
    const float* W = (const float*)d_in[1];   // 512 x 256
    const float* a = (const float*)d_in[2];   // 512 (a1 | a2)
    float* out = (float*)d_out;               // 8192 x 256

    char* ws = (char*)d_ws;
    _Float16*       hT   = (_Float16*)ws;                          // 4 MB
    float*          fbuf = (float*)(ws + 4194304);                 // 32 KB
    float*          gbuf = (float*)(ws + 4194304 + 32768);         // 32 KB
    float*          gmax = (float*)(ws + 4194304 + 65536);         // 4 B
    unsigned short* Wt   = (unsigned short*)(ws + 4194304 + 65536 + 64); // 256 KB

    k_wt  <<<512, 256, 0, stream>>>(W, Wt);
    k_h   <<<64,  256, 0, stream>>>(x, Wt, a, hT, fbuf, gbuf);
    k_gmax<<<1,   256, 0, stream>>>(gbuf, gmax);
    k_attn<<<64,  256, 0, stream>>>(hT, fbuf, gbuf, gmax, out);
}
